// ISTFT_80745385164952
// MI455X (gfx1250) — matrix operand order repaired and checked by reading, not compiled, not run on hardware
//
#include <hip/hip_runtime.h>
#include <hip/hip_bf16.h>

// ---- problem constants (from reference) ----
#define N_FFT   1024
#define HOP     256
#define NB      8
#define TFRM    2048
#define FREQ    513
#define OUTLEN  525312          // TFRM*HOP + N_FFT
#define OUTCROP 524288          // OUTLEN - N_FFT
#define KHALF   544             // 513 padded to multiple of 32
#define KP      1088            // 2*KHALF, padded GEMM K
#define KPAD    (KP + 8)        // LDS row stride in f16 (bank-conflict-free)
#define MTOT    (NB * TFRM)     // 16384 GEMM rows

typedef __attribute__((ext_vector_type(16))) _Float16 v16h;
typedef __attribute__((ext_vector_type(8)))  _Float16 v8h;
typedef __attribute__((ext_vector_type(8)))  float    v8f;

// ---------------------------------------------------------------------------
// 1) Stage X = [fr(513) | 0-pad | fi(513) | 0-pad] as f16, row-major K=1088.
// ---------------------------------------------------------------------------
__global__ void pack_x_kernel(const float* __restrict__ fr,
                              const float* __restrict__ fi,
                              _Float16* __restrict__ X) {
  const int k = blockIdx.x * 64 + threadIdx.x;
  const int r = blockIdx.y;                       // r = b*TFRM + t
  float v = 0.0f;
  if (k < FREQ)                          v = fr[(size_t)r * FREQ + k];
  else if (k >= KHALF && k < KHALF+FREQ) v = fi[(size_t)r * FREQ + (k - KHALF)];
  X[(size_t)r * KP + k] = (_Float16)v;
}

// ---------------------------------------------------------------------------
// 2) Combined synthesis weights, TRANSPOSED: Wt[n][k] (f16).
//    Wt[n][k] = c_k*W_real[k][n];  Wt[n][KHALF+k] = -c_k*W_imag[k][n]
//    c_k = 1 for k==0 or k==512, else 2 (Hermitian fold).
// ---------------------------------------------------------------------------
__global__ void pack_w_kernel(const float* __restrict__ Wr,
                              const float* __restrict__ Wi,
                              _Float16* __restrict__ Wt) {
  const int k = blockIdx.x * 64 + threadIdx.x;    // 0..1087
  const int n = blockIdx.y;                       // 0..1023
  float v = 0.0f;
  if (k < FREQ) {
    const float c = (k == 0 || k == 512) ? 1.0f : 2.0f;
    v = c * Wr[(size_t)k * N_FFT + n];
  } else if (k >= KHALF && k < KHALF + FREQ) {
    const int k2 = k - KHALF;
    const float c = (k2 == 0 || k2 == 512) ? 1.0f : 2.0f;
    v = -c * Wi[(size_t)k2 * N_FFT + n];
  }
  Wt[(size_t)n * KP + k] = (_Float16)v;
}

// ---------------------------------------------------------------------------
// 3) Zero OLA accumulator (NB * OUTLEN floats). 4,202,496 = 16416 * 256.
// ---------------------------------------------------------------------------
__global__ void zero_y_kernel(float* __restrict__ y) {
  y[(size_t)blockIdx.x * 256 + threadIdx.x] = 0.0f;
}

// ---------------------------------------------------------------------------
// 4) GEMM + fused overlap-add.
//    Block = 8 waves sharing one 16-row A tile (M0), each wave owns 16x64 N.
//    A tile async-copied to LDS once per block (ASYNCcnt path); rows padded
//    so ds_load_b128 A-fragment reads are bank-conflict free.
//    B fragments stream from L2 through a ping-pong register buffer with an
//    UNCONDITIONAL reload (straight-line loop body -> precise partial
//    s_wait_loadcnt instead of a full drain). The final iteration over-reads
//    <=128B past Wt, which lands in a dedicated pad region of the workspace.
//    Epilogue scatters into y[b, t*HOP + n] with global f32 atomics (OLA).
// ---------------------------------------------------------------------------
__global__ __launch_bounds__(256)
void istft_gemm_kernel(const _Float16* __restrict__ X,
                       const _Float16* __restrict__ Wt,
                       float* __restrict__ y) {
  __shared__ _Float16 sA[16 * KPAD];          // 35,072 B

  const int tid  = threadIdx.x;
  const int lane = tid & 31;
  const int wave = tid >> 5;
  const int gw   = blockIdx.x * 8 + wave;     // 262144 waves total
  const int M0   = (gw >> 4) * 16;            // block-uniform (8 waves share)
  const int N0   = (gw & 15) * 64;

  // ---- async-stage the 16xKP A tile into LDS (row-padded) ----
  {
    const char* gsrc  = (const char*)(X + (size_t)M0 * KP);
    const unsigned lbase = (unsigned)(uintptr_t)(void*)sA;
    // 16 rows * 136 16B-chunks/row = 2176 chunks, 256 threads
    for (int c = tid; c < 16 * (KP / 8); c += 256) {
      const int row = c / (KP / 8);
      const int col = c % (KP / 8);
      unsigned loff = lbase + (unsigned)(row * (KPAD * 2) + col * 16);
      unsigned long long gaddr =
          (unsigned long long)(uintptr_t)(gsrc + row * (KP * 2) + col * 16);
      asm volatile("global_load_async_to_lds_b128 %0, %1, off"
                   :: "v"(loff), "v"(gaddr) : "memory");
    }
    asm volatile("s_wait_asynccnt 0x0" ::: "memory");
    __syncthreads();
  }

  // A-fragment addressing (f16 A 16x32): lane = M + 16*kh, K contiguous
  const int am  = lane & 15;
  const int akb = (lane >> 4) << 4;           // 0 or 16
  const _Float16* saRow = sA + am * KPAD + akb;

  // B-fragment addressing (f16 B 32x16): lane = N + 16*kh, K contiguous
  const int bn  = lane & 15;
  const int bko = (lane >> 4) << 4;           // 0 or 16
  const _Float16* bptr[4];
#pragma unroll
  for (int s = 0; s < 4; ++s)
    bptr[s] = Wt + (size_t)(N0 + s * 16 + bn) * KP + bko;

  v8f acc[4] = {};
  v16h b0[4], b1[4];

#pragma unroll
  for (int s = 0; s < 4; ++s) b0[s] = *(const v16h*)(bptr[s]);

  for (int k0 = 0; k0 < KP; k0 += 64) {
    // both A fragments for this iteration (LDS, cheap dscnt waits)
    union { v16h v; v8h h[2]; } a0, a1;
    a0.h[0] = *(const v8h*)(saRow + k0);
    a0.h[1] = *(const v8h*)(saRow + k0 + 8);
    a1.h[0] = *(const v8h*)(saRow + k0 + 32);
    a1.h[1] = *(const v8h*)(saRow + k0 + 40);

    // B for second half-step (in flight across first WMMA group)
#pragma unroll
    for (int s = 0; s < 4; ++s) b1[s] = *(const v16h*)(bptr[s] + k0 + 32);

#pragma unroll
    for (int s = 0; s < 4; ++s)
      acc[s] = __builtin_amdgcn_wmma_f32_16x16x32_f16(
          false, a0.v, false, b0[s], (short)0, acc[s], false, false);

    // B for next iteration's first half-step — UNCONDITIONAL (last iteration
    // over-reads <=128B into the workspace pad region; values never used)
#pragma unroll
    for (int s = 0; s < 4; ++s) b0[s] = *(const v16h*)(bptr[s] + k0 + 64);

#pragma unroll
    for (int s = 0; s < 4; ++s)
      acc[s] = __builtin_amdgcn_wmma_f32_16x16x32_f16(
          false, a1.v, false, b1[s], (short)0, acc[s], false, false);
  }

  // fused overlap-add: y[b, t*HOP + ncol] += frames[r, ncol]
  const int ch = lane >> 4;
  const int cn = lane & 15;
#pragma unroll
  for (int s = 0; s < 4; ++s) {
    const int ncol = N0 + s * 16 + cn;
#pragma unroll
    for (int v = 0; v < 8; ++v) {
      const int r  = M0 + v + 8 * ch;
      const int bb = r >> 11;                 // / TFRM
      const int t  = r & (TFRM - 1);
      atomicAdd(&y[(size_t)bb * OUTLEN + (size_t)t * HOP + ncol], acc[s][v]);
    }
  }
}

// ---------------------------------------------------------------------------
// 5) Normalize by analytically-computed window sum and crop n_fft/2 margins.
// ---------------------------------------------------------------------------
__global__ void finalize_kernel(const float* __restrict__ y,
                                const float* __restrict__ ola,
                                float* __restrict__ out) {
  const size_t p = (size_t)blockIdx.x * 256 + threadIdx.x;   // 0..NB*OUTCROP-1
  const int b = (int)(p / OUTCROP);
  const int i = (int)(p % OUTCROP) + (N_FFT / 2);
  const int d = i - (N_FFT - 1);
  int jmin = (d > 0) ? ((d + HOP - 1) >> 8) : 0;
  int jmax = i >> 8;
  if (jmax > TFRM - 1) jmax = TFRM - 1;
  float w = 0.0f;
  for (int j = jmin; j <= jmax; ++j) w += ola[i - j * HOP];
  w = fmaxf(w, 1e-11f);
  out[p] = y[(size_t)b * OUTLEN + i] / w;
}

// ---------------------------------------------------------------------------
extern "C" void kernel_launch(void* const* d_in, const int* in_sizes, int n_in,
                              void* d_out, int out_size, void* d_ws, size_t ws_size,
                              hipStream_t stream) {
  const float* fr  = (const float*)d_in[0];   // real_stft  (B,1,T,FREQ)
  const float* fi  = (const float*)d_in[1];   // imag_stft
  const float* Wr  = (const float*)d_in[2];   // W_real (1024,1024)
  const float* Wi  = (const float*)d_in[3];   // W_imag
  const float* ola = (const float*)d_in[4];   // ola_window (1024)
  float* out = (float*)d_out;

  // workspace layout (256B-aligned, with a 4KB pad after Wt for the
  // pipelined GEMM's unconditional tail over-read)
  char* ws = (char*)d_ws;
  const size_t xBytes  = (size_t)MTOT * KP * 2;     // 35,651,584
  const size_t wtBytes = (size_t)N_FFT * KP * 2;    //  2,228,224
  const size_t padB    = 4096;
  _Float16* X  = (_Float16*)ws;
  _Float16* Wt = (_Float16*)(ws + xBytes);
  float*    y  = (float*)(ws + xBytes + wtBytes + padB);          // 16,809,984

  pack_x_kernel<<<dim3(KP / 64, MTOT), 64, 0, stream>>>(fr, fi, X);
  pack_w_kernel<<<dim3(KP / 64, N_FFT), 64, 0, stream>>>(Wr, Wi, Wt);
  zero_y_kernel<<<(NB * OUTLEN) / 256, 256, 0, stream>>>(y);
  istft_gemm_kernel<<<(MTOT * 16) / 8, 256, 0, stream>>>(X, Wt, y);
  finalize_kernel<<<(NB * OUTCROP) / 256, 256, 0, stream>>>(y, ola, out);
}
